// DenseBlock_84920093377082
// MI455X (gfx1250) — matrix operand order repaired and checked by reading, not compiled, not run on hardware
//
#include <hip/hip_runtime.h>
#include <cstdint>
#include <cstddef>

// ---------------------------------------------------------------------------
// Types
// ---------------------------------------------------------------------------
typedef __attribute__((ext_vector_type(16))) __bf16 v16bf;
typedef __attribute__((ext_vector_type(8)))  float  v8f;

#define BB 8
#define HH 256
#define WW 256
#define CFEAT 144   // 12 + 4*32 = 140, padded to 144 (multiple of 16)

static __device__ __forceinline__ unsigned pack_bf16(float a, float b) {
    union { __bf16 h[2]; unsigned u; } p;
    p.h[0] = (__bf16)a; p.h[1] = (__bf16)b;
    return p.u;
}

// ---------------------------------------------------------------------------
// Kernel 1: stage x (fp32 NCHW) into feature buffer F (bf16 NHWC, stride 144)
// channels [0,12)
// ---------------------------------------------------------------------------
__global__ __launch_bounds__(256) void stage_x_kernel(
        const float* __restrict__ x, __bf16* __restrict__ F) {
    int gid = blockIdx.x * 256 + threadIdx.x;        // B*H*W = 524288
    int b   = gid >> 16;
    int rem = gid & 65535;
    const float* xb = x + (size_t)b * 12 * 65536 + rem;
    __bf16* fp = F + (size_t)gid * CFEAT;
#pragma unroll
    for (int c = 0; c < 12; c += 2) {
        float v0 = xb[(size_t)c * 65536];
        float v1 = xb[(size_t)(c + 1) * 65536];
        *(unsigned*)(fp + c) = pack_bf16(v0, v1);
    }
}

// ---------------------------------------------------------------------------
// Kernel 2: deformable conv (offset conv -> bilinear sample -> 32ch conv)
// writes lrelu(x1) to F channels [12,44)
// ---------------------------------------------------------------------------
__global__ __launch_bounds__(256) void deform_kernel(
        const float* __restrict__ x,
        const float* __restrict__ w_off,
        const float* __restrict__ b_off,
        const float* __restrict__ w1,
        __bf16* __restrict__ F) {
    __shared__ float swoff[18 * 12 * 9];
    __shared__ float sw1[32 * 12 * 9];
    __shared__ float sboff[18];
    int tid = threadIdx.x;
    for (int i = tid; i < 18 * 12 * 9; i += 256) swoff[i] = w_off[i];
    for (int i = tid; i < 32 * 12 * 9; i += 256) sw1[i] = w1[i];
    if (tid < 18) sboff[tid] = b_off[tid];
    __syncthreads();

    int gid = blockIdx.x * 256 + tid;
    int b   = gid >> 16;
    int rem = gid & 65535;
    int y   = rem >> 8;
    int xx  = rem & 255;
    const float* xb = x + (size_t)b * 12 * 65536;

    // ---- offset conv: 18 outputs ----
    float off[18];
#pragma unroll
    for (int k = 0; k < 18; ++k) off[k] = sboff[k];
#pragma unroll
    for (int ky = 0; ky < 3; ++ky) {
#pragma unroll
        for (int kx = 0; kx < 3; ++kx) {
            int iy = y + ky - 1, ix = xx + kx - 1;
            if (iy < 0 || iy >= HH || ix < 0 || ix >= WW) continue;
            int tap = ky * 3 + kx;
#pragma unroll 1
            for (int c = 0; c < 12; ++c) {
                float xv = xb[(size_t)c * 65536 + iy * WW + ix];
#pragma unroll
                for (int k2 = 0; k2 < 18; ++k2)
                    off[k2] += xv * swoff[(k2 * 12 + c) * 9 + tap];
            }
        }
    }

    // ---- bilinear sample + 32-channel conv ----
    float acc[32];
#pragma unroll
    for (int m = 0; m < 32; ++m) acc[m] = 0.f;

#pragma unroll
    for (int tap = 0; tap < 9; ++tap) {
        int ky = tap / 3, kx = tap % 3;
        float py = (float)(y + ky - 1) + off[tap * 2 + 0];
        float px = (float)(xx + kx - 1) + off[tap * 2 + 1];
        float fy = floorf(py), fx = floorf(px);
        float ly = py - fy, lx = px - fx;
        int y0 = (int)fy, x0 = (int)fx;
        int y1 = y0 + 1,  x1 = x0 + 1;
        float vy0 = (y0 >= 0 && y0 < HH) ? 1.f : 0.f;
        float vy1 = (y1 >= 0 && y1 < HH) ? 1.f : 0.f;
        float vx0 = (x0 >= 0 && x0 < WW) ? 1.f : 0.f;
        float vx1 = (x1 >= 0 && x1 < WW) ? 1.f : 0.f;
        float w00 = (1.f - ly) * (1.f - lx) * vy0 * vx0;
        float w01 = (1.f - ly) * lx         * vy0 * vx1;
        float w10 = ly * (1.f - lx)         * vy1 * vx0;
        float w11 = ly * lx                 * vy1 * vx1;
        int yc0 = min(max(y0, 0), HH - 1), yc1 = min(max(y1, 0), HH - 1);
        int xc0 = min(max(x0, 0), WW - 1), xc1 = min(max(x1, 0), WW - 1);
        int i00 = yc0 * WW + xc0, i01 = yc0 * WW + xc1;
        int i10 = yc1 * WW + xc0, i11 = yc1 * WW + xc1;
#pragma unroll 1
        for (int c = 0; c < 12; ++c) {
            const float* xc = xb + (size_t)c * 65536;
            float s = w00 * xc[i00] + w01 * xc[i01]
                    + w10 * xc[i10] + w11 * xc[i11];
#pragma unroll
            for (int m = 0; m < 32; ++m)
                acc[m] += s * sw1[(m * 12 + c) * 9 + tap];
        }
    }

    __bf16* fp = F + (size_t)gid * CFEAT + 12;
#pragma unroll
    for (int m = 0; m < 32; m += 2) {
        float v0 = acc[m];     v0 = (v0 >= 0.f) ? v0 : 0.2f * v0;
        float v1 = acc[m + 1]; v1 = (v1 >= 0.f) ? v1 : 0.2f * v1;
        *(unsigned*)(fp + m) = pack_bf16(v0, v1);
    }
}

// ---------------------------------------------------------------------------
// Kernel 3: pack OIHW fp32 weights into WMMA A-fragment layout (bf16).
// Fragment element e of lane l (half = l>>4) maps to
//   Klocal = e + half*16   (contiguous K order, matching the B fragment reads)
// K = tap*CinP + c (tap-major, CinP padded channels).
// ---------------------------------------------------------------------------
__global__ __launch_bounds__(128) void pack_weights_kernel(
        const float* __restrict__ w, __bf16* __restrict__ wpack,
        int Cout, int Cin, int CinP, int nChunks, int mtiles) {
    int tid = blockIdx.x * 128 + threadIdx.x;
    int total = nChunks * mtiles * 32;
    if (tid >= total) return;
    int q = tid / (mtiles * 32);
    int r = tid - q * (mtiles * 32);
    int t = r >> 5;
    int l = r & 31;
    int m = t * 16 + (l & 15);
    int half = l >> 4;
    int Ktot = 9 * CinP;
    v16bf frag;
#pragma unroll
    for (int e = 0; e < 16; ++e) {
        int Klocal = e + half * 16;
        int K = q * 32 + Klocal;
        float val = 0.f;
        if (K < Ktot) {
            int tap = K / CinP;
            int c   = K - tap * CinP;
            if (c < Cin && m < Cout)
                val = w[((size_t)m * Cin + c) * 9 + tap];
        }
        frag[e] = (__bf16)val;
    }
    *(v16bf*)(wpack + (size_t)tid * 16) = frag;
}

// ---------------------------------------------------------------------------
// Kernel 4: dense 3x3 conv as implicit GEMM with v_wmma_f32_16x16x32_bf16.
// Block = 64 threads (2 waves), covers 64 pixels of one row of one image.
// Each wave computes 32 pixels (two N-tiles sharing A fragments) x
// (MTILES*16) output channels -> 2*MTILES WMMAs per K-chunk per wave.
// LDS halo tile (3 x 66 x CinP bf16, +16-elem zero pad for the K tail) is
// filled with GLOBAL_LOAD_ASYNC_TO_LDS_B64 (ASYNCcnt) where in-bounds.
// ---------------------------------------------------------------------------
template <int CIN, int CINP, int COUT, int MTILES, int NCHUNKS, int COFS,
          bool RELU, bool TO_F>
__global__ __launch_bounds__(64) void conv3x3_wmma_kernel(
        const __bf16* __restrict__ Fin,
        __bf16* __restrict__ Fout,
        const __bf16* __restrict__ wpack,
        const float* __restrict__ bias,
        float* __restrict__ out) {
    constexpr int ZOFF = 3 * 66 * CINP;
    __shared__ __align__(32) __bf16 smem[ZOFF + 16];

    int tid  = threadIdx.x;
    int blk  = blockIdx.x;
    int tile = blk & 3;
    int t2   = blk >> 2;
    int h    = t2 & 255;
    int b    = t2 >> 8;
    int w0   = tile * 64;

    // ---- stage halo tile: 3 rows x 66 cols x CIN ch (pad to CINP, zero OOB)
    for (int pos = tid; pos < 3 * 66; pos += 64) {
        int rrow = pos / 66;
        int col  = pos - rrow * 66;
        int gy = h + rrow - 1;
        int gx = w0 + col - 1;
        if (gy >= 0 && gy < HH && gx >= 0 && gx < WW) {
            const __bf16* src = Fin + ((size_t)(b * HH + gy) * WW + gx) * CFEAT;
            uint32_t lbase = (uint32_t)(uintptr_t)(smem + (size_t)pos * CINP);
#pragma unroll
            for (int i = 0; i < CIN / 4; ++i) {
                uint64_t ga = (uint64_t)(uintptr_t)(src + i * 4);
                uint32_t la = lbase + i * 8;
                asm volatile("global_load_async_to_lds_b64 %0, %1, off"
                             :: "v"(la), "v"(ga) : "memory");
            }
            // zero the 4 padded channels (CINP-CIN == 4) with a normal DS store
            ((uint2*)(smem + (size_t)pos * CINP))[CIN / 4] = make_uint2(0u, 0u);
        } else {
            uint2* dst = (uint2*)(smem + (size_t)pos * CINP);
#pragma unroll
            for (int i = 0; i < CINP / 4; ++i) dst[i] = make_uint2(0u, 0u);
        }
    }
    if (tid < 16) smem[ZOFF + tid] = (__bf16)0.0f;      // K-tail zero fragment
    asm volatile("s_wait_asynccnt 0x0" ::: "memory");
    __syncthreads();

    int lane = tid & 31;
    int wv   = tid >> 5;                                // 0..1
    int half = lane >> 4;
    int colB = wv * 32 + (lane & 15);                   // staging col base (dx=0)

    v8f acc[MTILES][2] = {};
    int t0 = 0, c0 = 0;                                 // (tap, ch) for K = q*32
    for (int q = 0; q < NCHUNKS; ++q) {
        int t1 = t0, c1 = c0 + 16;
        if (c1 >= CINP) { c1 -= CINP; t1 += 1; }
        int tt = half ? t1 : t0;
        int cc = half ? c1 : c0;
        int dy = tt / 3;
        int dx = tt - dy * 3;
        int base0 = (dy * 66 + colB + dx) * CINP + cc;
        int a0 = (tt < 9) ? base0 : ZOFF;
        int a1 = (tt < 9) ? (base0 + 16 * CINP) : ZOFF;
        v16bf bf0 = *(const v16bf*)(smem + a0);
        v16bf bf1 = *(const v16bf*)(smem + a1);
#pragma unroll
        for (int mt = 0; mt < MTILES; ++mt) {
            v16bf afrag = *(const v16bf*)(wpack +
                (((size_t)q * MTILES + mt) * 32 + lane) * 16);
            acc[mt][0] = __builtin_amdgcn_wmma_f32_16x16x32_bf16(
                false, afrag, false, bf0, (short)0, acc[mt][0], false, false);
            acc[mt][1] = __builtin_amdgcn_wmma_f32_16x16x32_bf16(
                false, afrag, false, bf1, (short)0, acc[mt][1], false, false);
        }
        c0 += 32;
        if (c0 >= CINP) { c0 -= CINP; t0 += 1; }
    }

    // ---- epilogue: bias (+lrelu), store ----
#pragma unroll
    for (int g = 0; g < 2; ++g) {
        int px = w0 + wv * 32 + g * 16 + (lane & 15);
        size_t pix = (size_t)(b * HH + h) * WW + px;
#pragma unroll
        for (int mt = 0; mt < MTILES; ++mt) {
#pragma unroll
            for (int r = 0; r < 8; r += 2) {
                int m = mt * 16 + half * 8 + r;
                float v0 = acc[mt][g][r]     + ((m     < COUT) ? bias[m]     : 0.f);
                float v1 = acc[mt][g][r + 1] + ((m + 1 < COUT) ? bias[m + 1] : 0.f);
                if (RELU) {
                    v0 = (v0 >= 0.f) ? v0 : 0.2f * v0;
                    v1 = (v1 >= 0.f) ? v1 : 0.2f * v1;
                }
                if (TO_F) {
                    *(unsigned*)(Fout + pix * CFEAT + COFS + m) = pack_bf16(v0, v1);
                } else {
                    if (m < COUT)
                        out[((size_t)(b * COUT + m) * HH + h) * WW + px] = v0;
                    if (m + 1 < COUT)
                        out[((size_t)(b * COUT + m + 1) * HH + h) * WW + px] = v1;
                }
            }
        }
    }
}

// ---------------------------------------------------------------------------
// Host launcher
// ---------------------------------------------------------------------------
static inline size_t align256(size_t v) { return (v + 255) & ~(size_t)255; }

extern "C" void kernel_launch(void* const* d_in, const int* in_sizes, int n_in,
                              void* d_out, int out_size, void* d_ws, size_t ws_size,
                              hipStream_t stream) {
    const float* x     = (const float*)d_in[0];
    const float* w_off = (const float*)d_in[1];
    const float* b_off = (const float*)d_in[2];
    const float* w1    = (const float*)d_in[3];
    const float* w2    = (const float*)d_in[4];
    const float* b2    = (const float*)d_in[5];
    const float* w3    = (const float*)d_in[6];
    const float* b3    = (const float*)d_in[7];
    const float* w4    = (const float*)d_in[8];
    const float* b4    = (const float*)d_in[9];
    const float* w5    = (const float*)d_in[10];
    const float* b5    = (const float*)d_in[11];
    float* out = (float*)d_out;

    char* ws = (char*)d_ws;
    size_t off = 0;
    __bf16* F = (__bf16*)(ws + off);
    off += align256((size_t)BB * HH * WW * CFEAT * 2);
    __bf16* wp2 = (__bf16*)(ws + off); off += align256((size_t)14 * 2 * 32 * 16 * 2);
    __bf16* wp3 = (__bf16*)(ws + off); off += align256((size_t)23 * 2 * 32 * 16 * 2);
    __bf16* wp4 = (__bf16*)(ws + off); off += align256((size_t)32 * 2 * 32 * 16 * 2);
    __bf16* wp5 = (__bf16*)(ws + off); off += align256((size_t)41 * 1 * 32 * 16 * 2);
    (void)ws_size; (void)in_sizes; (void)n_in; (void)out_size;

    const int nPix = BB * HH * WW;                 // 524288

    // weight packing (tiny)
    pack_weights_kernel<<<(14 * 2 * 32 + 127) / 128, 128, 0, stream>>>(
        w2, wp2, 32, 44, 48, 14, 2);
    pack_weights_kernel<<<(23 * 2 * 32 + 127) / 128, 128, 0, stream>>>(
        w3, wp3, 32, 76, 80, 23, 2);
    pack_weights_kernel<<<(32 * 2 * 32 + 127) / 128, 128, 0, stream>>>(
        w4, wp4, 32, 108, 112, 32, 2);
    pack_weights_kernel<<<(41 * 1 * 32 + 127) / 128, 128, 0, stream>>>(
        w5, wp5, 12, 140, 144, 41, 1);

    // stage x into F channels [0,12)
    stage_x_kernel<<<nPix / 256, 256, 0, stream>>>(x, F);

    // deformable conv -> F channels [12,44)
    deform_kernel<<<nPix / 256, 256, 0, stream>>>(x, w_off, b_off, w1, F);

    const int convGrid = BB * HH * (WW / 64);      // 8192 blocks, 64 threads
    // conv2: 44 -> 32, lrelu, write F[44,76)
    conv3x3_wmma_kernel<44, 48, 32, 2, 14, 44, true, true>
        <<<convGrid, 64, 0, stream>>>(F, F, wp2, b2, nullptr);
    // conv3: 76 -> 32, lrelu, write F[76,108)
    conv3x3_wmma_kernel<76, 80, 32, 2, 23, 76, true, true>
        <<<convGrid, 64, 0, stream>>>(F, F, wp3, b3, nullptr);
    // conv4: 108 -> 32, lrelu, write F[108,140)
    conv3x3_wmma_kernel<108, 112, 32, 2, 32, 108, true, true>
        <<<convGrid, 64, 0, stream>>>(F, F, wp4, b4, nullptr);
    // conv5: 140 -> 12, no lrelu, write fp32 NCHW to d_out
    conv3x3_wmma_kernel<140, 144, 12, 1, 41, 0, false, false>
        <<<convGrid, 64, 0, stream>>>(F, F, wp5, b5, out);
}
